// Glom_41738492183069
// MI455X (gfx1250) — matrix operand order repaired and checked by reading, not compiled, not run on hardware
//
#include <hip/hip_runtime.h>
#include <math.h>

// ---------------------------------------------------------------------------
// GLOM forward for MI455X (gfx1250, wave32, WMMA).
// All GEMM work uses v_wmma_f32_16x16x32_bf16 (f32 accumulate).
// B=8, S=224, P=14, D=512, L=6, NPS=16, N=256, G=5, H=2048, iters=12.
// Working set (bf16 weights 42MB + activations) is L2-resident (192MB), so
// the kernel is matrix-pipe bound: staging is vectorized (b128/b64) and
// software-pipelined so global loads overlap WMMA.
// ---------------------------------------------------------------------------

#define BM 128
#define BN 128
#define BK 32
#define LSTR 40   // LDS tile stride in halfwords: 80B row, 16B-aligned, bank-friendly

typedef __attribute__((ext_vector_type(16))) __bf16 v16bf;
typedef __attribute__((ext_vector_type(8)))  float  v8f;

union FragBF { v16bf v; unsigned u[8]; };

__device__ __forceinline__ unsigned short f2bf(float x) {
  union { float f; unsigned u; } c; c.f = x;
  unsigned r = c.u + 0x7FFFu + ((c.u >> 16) & 1u);  // round-to-nearest-even
  return (unsigned short)(r >> 16);
}
__device__ __forceinline__ unsigned pack2bf(float a, float b) {
  return (unsigned)f2bf(a) | ((unsigned)f2bf(b) << 16);
}

// A fragment: 16x32 bf16 tile, LDS row-major [m][k] (LSTR stride).
// ISA: lanes 0-15 row M=lane K={0..7,16..23}; lanes 16-31 same rows K={8..15,24..31}.
__device__ __forceinline__ void load_a_frag(const unsigned short* As, int m0, int lane, FragBF& f) {
  const int hf = lane >> 4;
  const int m  = lane & 15;
  const unsigned short* row = As + (m0 + m) * LSTR;
#pragma unroll
  for (int j = 0; j < 8; ++j) {
    const int k = ((j >> 2) << 4) + (hf << 3) + ((j & 3) << 1);
    f.u[j] = *(const unsigned*)(row + k);
  }
}

// B fragment: 32x16 bf16 tile, LDS stored transposed [n][k] (LSTR stride).
// ISA: lanes 0-15 col N=lane K={0..7,16..23}; lanes 16-31 col N=lane-16 K={8..15,24..31}.
__device__ __forceinline__ void load_b_frag(const unsigned short* Bs, int n0, int lane, FragBF& f) {
  const int hf = lane >> 4;
  const int n  = lane & 15;
  const unsigned short* col = Bs + (n0 + n) * LSTR + (hf << 3);
#pragma unroll
  for (int j = 0; j < 8; ++j) f.u[j] = *(const unsigned*)(col + ((j >> 2) << 4) + ((j & 3) << 1));
}

#define WAVE_COORDS \
  const int tid  = threadIdx.x; \
  const int lane = tid & 31;    \
  const int wid  = tid >> 5;    \
  const int wm   = (wid >> 1) * 32; \
  const int wn   = (wid & 1) * 64;

#define ACC_ZERO \
  v8f acc[2][4]; \
  _Pragma("unroll") for (int mi = 0; mi < 2; ++mi) \
  _Pragma("unroll") for (int ni = 0; ni < 4; ++ni) acc[mi][ni] = 0.0f;

// 2-stage software pipeline: regs -> LDS -> WMMA, next-slab global loads
// issued before the MMA block so they overlap the matrix math.
template <class LA, class SA, class LB, class SB>
__device__ __forceinline__ void gemm_pipeline(
    int KTOT, LA&& loadA, SA&& storeA, LB&& loadB, SB&& storeB,
    const unsigned short* As, const unsigned short* Bs,
    v8f (&acc)[2][4], int wm, int wn, int lane) {
  loadA(0); loadB(0);
  for (int kt = 0; kt < KTOT; kt += BK) {
    storeA(); storeB();
    __syncthreads();
    if (kt + BK < KTOT) { loadA(kt + BK); loadB(kt + BK); }
    FragBF afr[2]; FragBF bfr[4];
    load_a_frag(As, wm,      lane, afr[0]);
    load_a_frag(As, wm + 16, lane, afr[1]);
#pragma unroll
    for (int ni = 0; ni < 4; ++ni) load_b_frag(Bs, wn + ni * 16, lane, bfr[ni]);
#pragma unroll
    for (int mi = 0; mi < 2; ++mi)
#pragma unroll
      for (int ni = 0; ni < 4; ++ni)
        acc[mi][ni] = __builtin_amdgcn_wmma_f32_16x16x32_bf16(
            false, afr[mi].v, false, bfr[ni].v, (short)0, acc[mi][ni], false, false);
    __syncthreads();
  }
}

// C/D layout: VGPR r -> lanes 0-15 row r, lanes 16-31 row r+8; col = lane&15.
#define EPILOGUE_LOOP(...) \
  { const int hf = lane >> 4; const int lcol = lane & 15; \
    _Pragma("unroll") for (int mi = 0; mi < 2; ++mi) \
    _Pragma("unroll") for (int ni = 0; ni < 4; ++ni) \
    _Pragma("unroll") for (int r = 0; r < 8; ++r) { \
      const int row = tileM + wm + mi * 16 + r + (hf << 3); \
      const int col = tileN + wn + ni * 16 + lcol; \
      const float accv = acc[mi][ni][r]; \
      __VA_ARGS__ } }

// ---------------------------------------------------------------------------
// FFN GEMM1: H = GELU(X @ W1[l] + b1[l]) -> bf16. M=2048, N=2048, K=512.
// X rows from lv[(m, level = z + LVLOFF)], compile-time +pos_emb.
// ---------------------------------------------------------------------------
template <int LVLOFF, bool ADDPOS>
__global__ void __launch_bounds__(256) glom_ff_gemm1(
    const float* __restrict__ lv, const float* __restrict__ pos,
    const unsigned short* __restrict__ W1, const float* __restrict__ b1,
    unsigned short* __restrict__ Hbuf) {
  __shared__ __align__(16) unsigned short As[BM * LSTR];
  __shared__ __align__(16) unsigned short Bs[BN * LSTR];
  WAVE_COORDS
  const int tileN = blockIdx.x * BN;
  const int tileM = blockIdx.y * BM;
  const int lvl   = blockIdx.z;
  const unsigned short* Wl = W1 + (size_t)lvl * 512 * 2048;
  ACC_ZERO
  float4 areg[4];
  uint2  breg[4];
  auto loadA = [&](int kt) {
#pragma unroll
    for (int s = 0; s < 4; ++s) {
      const int idx4 = s * 256 + tid;
      const int row = idx4 >> 3;
      const int kb  = (idx4 & 7) * 4;
      const int m = tileM + row;
      float4 v = *(const float4*)(lv + ((size_t)m * 6 + lvl + LVLOFF) * 512 + kt + kb);
      if (ADDPOS) {
        float4 w = *(const float4*)(pos + (size_t)(m & 255) * 512 + kt + kb);
        v.x += w.x; v.y += w.y; v.z += w.z; v.w += w.w;
      }
      areg[s] = v;
    }
  };
  auto storeA = [&]() {
#pragma unroll
    for (int s = 0; s < 4; ++s) {
      const int idx4 = s * 256 + tid;
      const int row = idx4 >> 3;
      const int kb  = (idx4 & 7) * 4;
      uint2 p; p.x = pack2bf(areg[s].x, areg[s].y); p.y = pack2bf(areg[s].z, areg[s].w);
      *(uint2*)&As[row * LSTR + kb] = p;
    }
  };
  auto loadB = [&](int kt) {
#pragma unroll
    for (int s = 0; s < 4; ++s) {
      const int idxv = s * 256 + tid;
      const int k  = idxv >> 5;
      const int nb = (idxv & 31) * 4;
      breg[s] = *(const uint2*)(Wl + (size_t)(kt + k) * 2048 + tileN + nb);
    }
  };
  auto storeB = [&]() {
#pragma unroll
    for (int s = 0; s < 4; ++s) {
      const int idxv = s * 256 + tid;
      const int k  = idxv >> 5;
      const int nb = (idxv & 31) * 4;
      Bs[(nb + 0) * LSTR + k] = (unsigned short)(breg[s].x);
      Bs[(nb + 1) * LSTR + k] = (unsigned short)(breg[s].x >> 16);
      Bs[(nb + 2) * LSTR + k] = (unsigned short)(breg[s].y);
      Bs[(nb + 3) * LSTR + k] = (unsigned short)(breg[s].y >> 16);
    }
  };
  gemm_pipeline(512, loadA, storeA, loadB, storeB, As, Bs, acc, wm, wn, lane);
  EPILOGUE_LOOP(
    float h = accv + b1[lvl * 2048 + col];
    float g = 0.5f * h * (1.0f + erff(h * 0.70710678118654752f));
    Hbuf[((size_t)lvl * 2048 + row) * 2048 + col] = f2bf(g);
  )
}

// ---------------------------------------------------------------------------
// FFN GEMM2: Out = H @ W2[l] + b2[l] (f32). M=2048, N=512, K=2048.
// Out layout: [m][level(5)][512].
// ---------------------------------------------------------------------------
__global__ void __launch_bounds__(256) glom_ff_gemm2(
    const unsigned short* __restrict__ Hbuf, const unsigned short* __restrict__ W2,
    const float* __restrict__ b2, float* __restrict__ outp) {
  __shared__ __align__(16) unsigned short As[BM * LSTR];
  __shared__ __align__(16) unsigned short Bs[BN * LSTR];
  WAVE_COORDS
  const int tileN = blockIdx.x * BN;
  const int tileM = blockIdx.y * BM;
  const int lvl   = blockIdx.z;
  const unsigned short* Hl = Hbuf + (size_t)lvl * 2048 * 2048;
  const unsigned short* Wl = W2   + (size_t)lvl * 2048 * 512;
  ACC_ZERO
  uint4 areg[2];
  uint2 breg[4];
  auto loadA = [&](int kt) {
#pragma unroll
    for (int s = 0; s < 2; ++s) {
      const int idx8 = s * 256 + tid;
      const int row = idx8 >> 2;
      const int kb  = (idx8 & 3) * 8;
      areg[s] = *(const uint4*)(Hl + (size_t)(tileM + row) * 2048 + kt + kb);
    }
  };
  auto storeA = [&]() {
#pragma unroll
    for (int s = 0; s < 2; ++s) {
      const int idx8 = s * 256 + tid;
      const int row = idx8 >> 2;
      const int kb  = (idx8 & 3) * 8;
      *(uint4*)&As[row * LSTR + kb] = areg[s];
    }
  };
  auto loadB = [&](int kt) {
#pragma unroll
    for (int s = 0; s < 4; ++s) {
      const int idxv = s * 256 + tid;
      const int k  = idxv >> 5;
      const int nb = (idxv & 31) * 4;
      breg[s] = *(const uint2*)(Wl + (size_t)(kt + k) * 512 + tileN + nb);
    }
  };
  auto storeB = [&]() {
#pragma unroll
    for (int s = 0; s < 4; ++s) {
      const int idxv = s * 256 + tid;
      const int k  = idxv >> 5;
      const int nb = (idxv & 31) * 4;
      Bs[(nb + 0) * LSTR + k] = (unsigned short)(breg[s].x);
      Bs[(nb + 1) * LSTR + k] = (unsigned short)(breg[s].x >> 16);
      Bs[(nb + 2) * LSTR + k] = (unsigned short)(breg[s].y);
      Bs[(nb + 3) * LSTR + k] = (unsigned short)(breg[s].y >> 16);
    }
  };
  gemm_pipeline(2048, loadA, storeA, loadB, storeB, As, Bs, acc, wm, wn, lane);
  EPILOGUE_LOOP(
    outp[((size_t)row * 5 + lvl) * 512 + col] = accv + b2[lvl * 512 + col];
  )
}

// ---------------------------------------------------------------------------
// Consensus sim GEMM: sim[z] = Lv[z] @ K[z]^T * D^-0.5, diag = -0.0005.
// z = b*6+l (48). M=256, N=256, K=512.
// ---------------------------------------------------------------------------
__global__ void __launch_bounds__(256) glom_sim_gemm(
    const unsigned short* __restrict__ lvbf, const unsigned short* __restrict__ kbf,
    float* __restrict__ sim) {
  __shared__ __align__(16) unsigned short As[BM * LSTR];
  __shared__ __align__(16) unsigned short Bs[BN * LSTR];
  WAVE_COORDS
  const int tileN = blockIdx.x * BN;
  const int tileM = blockIdx.y * BM;
  const int z     = blockIdx.z;
  const unsigned short* A  = lvbf + (size_t)z * 256 * 512;
  const unsigned short* Bm = kbf  + (size_t)z * 256 * 512;
  ACC_ZERO
  uint4 areg[2];
  uint2 breg[4];
  auto loadA = [&](int kt) {
#pragma unroll
    for (int s = 0; s < 2; ++s) {
      const int idx8 = s * 256 + tid;
      const int row = idx8 >> 2;
      const int kb  = (idx8 & 3) * 8;
      areg[s] = *(const uint4*)(A + (size_t)(tileM + row) * 512 + kt + kb);
    }
  };
  auto storeA = [&]() {
#pragma unroll
    for (int s = 0; s < 2; ++s) {
      const int idx8 = s * 256 + tid;
      const int row = idx8 >> 2;
      const int kb  = (idx8 & 3) * 8;
      *(uint4*)&As[row * LSTR + kb] = areg[s];
    }
  };
  // B^T is given directly (K[n][k]): k-contiguous reads, vector LDS stores.
  auto loadB = [&](int kt) {
#pragma unroll
    for (int s = 0; s < 4; ++s) {
      const int idxv = s * 256 + tid;
      const int n  = idxv >> 3;
      const int kb = (idxv & 7) * 4;
      breg[s] = *(const uint2*)(Bm + (size_t)(tileN + n) * 512 + kt + kb);
    }
  };
  auto storeB = [&]() {
#pragma unroll
    for (int s = 0; s < 4; ++s) {
      const int idxv = s * 256 + tid;
      const int n  = idxv >> 3;
      const int kb = (idxv & 7) * 4;
      *(uint2*)&Bs[n * LSTR + kb] = breg[s];
    }
  };
  gemm_pipeline(512, loadA, storeA, loadB, storeB, As, Bs, acc, wm, wn, lane);
  EPILOGUE_LOOP(
    float s = accv * 0.04419417382415922f;   // 512^-0.5
    if (row == col) s = -0.0005f;            // TOKEN_ATTEND_SELF_VALUE
    sim[((size_t)z * 256 + row) * 256 + col] = s;
  )
}

// ---------------------------------------------------------------------------
// Consensus apply + fused combine:
// cons = attn[z] @ Lv[z]; new = (lv + bu_cat + td_pad + cons) / contrib.
// z = b*6+l. M=256 (tokens), N=512 (d), K=256.
// ---------------------------------------------------------------------------
__global__ void __launch_bounds__(256) glom_cons_gemm(
    const unsigned short* __restrict__ attn, const unsigned short* __restrict__ lvbf,
    const float* __restrict__ lv_in, const float* __restrict__ tokens,
    const float* __restrict__ bu_o, const float* __restrict__ td_o,
    float* __restrict__ lv_out) {
  __shared__ __align__(16) unsigned short As[BM * LSTR];
  __shared__ __align__(16) unsigned short Bs[BN * LSTR];
  WAVE_COORDS
  const int tileN = blockIdx.x * BN;
  const int tileM = blockIdx.y * BM;
  const int z     = blockIdx.z;
  const int bImg  = z / 6, l = z % 6;
  const unsigned short* A  = attn + (size_t)z * 256 * 256;
  const unsigned short* Bm = lvbf + (size_t)z * 256 * 512;
  ACC_ZERO
  uint4 areg[2];
  uint2 breg[4];
  auto loadA = [&](int kt) {
#pragma unroll
    for (int s = 0; s < 2; ++s) {
      const int idx8 = s * 256 + tid;
      const int row = idx8 >> 2;
      const int kb  = (idx8 & 3) * 8;
      areg[s] = *(const uint4*)(A + (size_t)(tileM + row) * 256 + kt + kb);
    }
  };
  auto storeA = [&]() {
#pragma unroll
    for (int s = 0; s < 2; ++s) {
      const int idx8 = s * 256 + tid;
      const int row = idx8 >> 2;
      const int kb  = (idx8 & 3) * 8;
      *(uint4*)&As[row * LSTR + kb] = areg[s];
    }
  };
  auto loadB = [&](int kt) {
#pragma unroll
    for (int s = 0; s < 4; ++s) {
      const int idxv = s * 256 + tid;
      const int k  = idxv >> 5;
      const int nb = (idxv & 31) * 4;
      breg[s] = *(const uint2*)(Bm + (size_t)(kt + k) * 512 + tileN + nb);
    }
  };
  auto storeB = [&]() {
#pragma unroll
    for (int s = 0; s < 4; ++s) {
      const int idxv = s * 256 + tid;
      const int k  = idxv >> 5;
      const int nb = (idxv & 31) * 4;
      Bs[(nb + 0) * LSTR + k] = (unsigned short)(breg[s].x);
      Bs[(nb + 1) * LSTR + k] = (unsigned short)(breg[s].x >> 16);
      Bs[(nb + 2) * LSTR + k] = (unsigned short)(breg[s].y);
      Bs[(nb + 3) * LSTR + k] = (unsigned short)(breg[s].y >> 16);
    }
  };
  gemm_pipeline(256, loadA, storeA, loadB, storeB, As, Bs, acc, wm, wn, lane);
  EPILOGUE_LOOP(
    const int m = bImg * 256 + row;
    const size_t oidx = ((size_t)m * 6 + l) * 512 + col;
    float lvold = lv_in[oidx];
    float buc = (l == 0) ? tokens[(size_t)m * 512 + col]
                         : bu_o[((size_t)m * 5 + (l - 1)) * 512 + col];
    float tdc = (l < 5) ? td_o[((size_t)m * 5 + l) * 512 + col] : 0.0f;
    float contrib = (l == 5) ? 3.0f : 4.0f;
    lv_out[oidx] = (lvold + buc + tdc + accv) / contrib;
  )
}

// ---------------------------------------------------------------------------
// Patch embedding: tokens[m][d] = patch(m) . W_patch[:,d] + b_patch[d].
// ---------------------------------------------------------------------------
__global__ void __launch_bounds__(256) glom_patch_embed(
    const float* __restrict__ img, const float* __restrict__ Wp,
    const float* __restrict__ bp, float* __restrict__ tokens) {
  const int m = blockIdx.x;
  const int b = m >> 8, n = m & 255;
  const int ny = n >> 4, nx = n & 15;
  __shared__ float xp[588];
  const int t = threadIdx.x;
  for (int i = t; i < 588; i += 256) {
    const int c = i % 3, pj = (i / 3) % 14, pi = i / 42;
    xp[i] = img[(((size_t)b * 3 + c) * 224 + (ny * 14 + pi)) * 224 + (nx * 14 + pj)];
  }
  __syncthreads();
  for (int d = t; d < 512; d += 256) {
    float s = bp[d];
    for (int k = 0; k < 588; ++k) s += xp[k] * Wp[(size_t)k * 512 + d];
    tokens[(size_t)m * 512 + d] = s;
  }
}

// f32 -> bf16 weight conversion (weights stay resident in L2 across iters)
__global__ void __launch_bounds__(256) glom_cvt_bf16(
    const float* __restrict__ src, unsigned short* __restrict__ dst, int n) {
  const int i = blockIdx.x * 256 + threadIdx.x;
  if (i < n) dst[i] = f2bf(src[i]);
}

// levels[b,n,l,d] = init_levels[l,d]
__global__ void __launch_bounds__(256) glom_init_levels(
    const float* __restrict__ initl, float* __restrict__ lv) {
  const size_t i = (size_t)blockIdx.x * 256 + threadIdx.x;
  if (i < 6291456ull) lv[i] = initl[i % 3072];
}

// Per-(b,l,n) row: bf16 copies of levels and L2-normalized levels (layout b,l,n,d).
__global__ void __launch_bounds__(256) glom_normalize(
    const float* __restrict__ lv, unsigned short* __restrict__ lvbf,
    unsigned short* __restrict__ kbf) {
  const int blk = blockIdx.x;          // ((b*6+l)*256 + n)
  const int z = blk >> 8, n = blk & 255;
  const int b = z / 6, l = z % 6;
  const int t = threadIdx.x;
  const float* src = lv + (((size_t)(b * 256 + n)) * 6 + l) * 512;
  const float x0 = src[t], x1 = src[t + 256];
  __shared__ float red[256];
  red[t] = x0 * x0 + x1 * x1;
  __syncthreads();
  for (int o = 128; o > 0; o >>= 1) { if (t < o) red[t] += red[t + o]; __syncthreads(); }
  const float inv = 1.0f / fmaxf(sqrtf(red[0]), 1e-12f);
  const size_t ob = (size_t)blk * 512;
  lvbf[ob + t]       = f2bf(x0);
  lvbf[ob + t + 256] = f2bf(x1);
  kbf[ob + t]        = f2bf(x0 * inv);
  kbf[ob + t + 256]  = f2bf(x1 * inv);
}

// Row softmax over 256 columns -> bf16 attn.
__global__ void __launch_bounds__(256) glom_softmax(
    const float* __restrict__ sim, unsigned short* __restrict__ attn) {
  const int row = blockIdx.x;
  const int t = threadIdx.x;
  const float v = sim[(size_t)row * 256 + t];
  __shared__ float red[256];
  red[t] = v;
  __syncthreads();
  for (int o = 128; o > 0; o >>= 1) { if (t < o) red[t] = fmaxf(red[t], red[t + o]); __syncthreads(); }
  const float mx = red[0];
  __syncthreads();
  const float e = __expf(v - mx);
  red[t] = e;
  __syncthreads();
  for (int o = 128; o > 0; o >>= 1) { if (t < o) red[t] += red[t + o]; __syncthreads(); }
  attn[(size_t)row * 256 + t] = f2bf(e / red[0]);
}

// ---------------------------------------------------------------------------
extern "C" void kernel_launch(void* const* d_in, const int* in_sizes, int n_in,
                              void* d_out, int out_size, void* d_ws, size_t ws_size,
                              hipStream_t stream) {
  (void)in_sizes; (void)n_in; (void)out_size; (void)ws_size;
  const float* img     = (const float*)d_in[0];
  const float* W_patch = (const float*)d_in[1];
  const float* b_patch = (const float*)d_in[2];
  const float* pos_emb = (const float*)d_in[3];
  const float* init_lv = (const float*)d_in[4];
  const float* bu_W1   = (const float*)d_in[5];
  const float* bu_b1   = (const float*)d_in[6];
  const float* bu_W2   = (const float*)d_in[7];
  const float* bu_b2   = (const float*)d_in[8];
  const float* td_W1   = (const float*)d_in[9];
  const float* td_b1   = (const float*)d_in[10];
  const float* td_W2   = (const float*)d_in[11];
  const float* td_b2   = (const float*)d_in[12];
  // d_in[13] = iters (device scalar); reference uses iters == 12.
  float* out = (float*)d_out;

  char* ws = (char*)d_ws;
  size_t off = 0;
  auto alloc = [&](size_t bytes) -> void* {
    void* p = ws + off;
    off = (off + bytes + 255) & ~(size_t)255;
    return p;
  };

  const size_t NW  = 5ull * 512 * 2048;        // grouped weight elems
  float*          tokens = (float*)alloc(2048ull * 512 * 4);
  float*          lvP0   = (float*)alloc(6291456ull * 4);
  float*          lvP1   = (float*)alloc(6291456ull * 4);
  unsigned short* w1bu   = (unsigned short*)alloc(NW * 2);
  unsigned short* w2bu   = (unsigned short*)alloc(NW * 2);
  unsigned short* w1td   = (unsigned short*)alloc(NW * 2);
  unsigned short* w2td   = (unsigned short*)alloc(NW * 2);
  unsigned short* Hbuf   = (unsigned short*)alloc(5ull * 2048 * 2048 * 2);
  float*          bu_o   = (float*)alloc(2048ull * 5 * 512 * 4);
  float*          td_o   = (float*)alloc(2048ull * 5 * 512 * 4);
  unsigned short* lvbf   = (unsigned short*)alloc(6291456ull * 2);
  unsigned short* kbf    = (unsigned short*)alloc(6291456ull * 2);
  float*          sim    = (float*)alloc(3145728ull * 4);
  unsigned short* attn   = (unsigned short*)alloc(3145728ull * 2);

  // One-shot setup
  glom_patch_embed<<<2048, 256, 0, stream>>>(img, W_patch, b_patch, tokens);
  glom_cvt_bf16<<<20480, 256, 0, stream>>>(bu_W1, w1bu, (int)NW);
  glom_cvt_bf16<<<20480, 256, 0, stream>>>(bu_W2, w2bu, (int)NW);
  glom_cvt_bf16<<<20480, 256, 0, stream>>>(td_W1, w1td, (int)NW);
  glom_cvt_bf16<<<20480, 256, 0, stream>>>(td_W2, w2td, (int)NW);
  glom_init_levels<<<24576, 256, 0, stream>>>(init_lv, lvP0);

  float* cur = lvP0;
  for (int it = 1; it <= 12; ++it) {
    float* nxt = (it == 12) ? out : ((cur == lvP0) ? lvP1 : lvP0);
    // bottom-up FFN: input = lv levels 0..4
    glom_ff_gemm1<0, false><<<dim3(16, 16, 5), 256, 0, stream>>>(cur, pos_emb, w1bu, bu_b1, Hbuf);
    glom_ff_gemm2<<<dim3(4, 16, 5), 256, 0, stream>>>(Hbuf, w2bu, bu_b2, bu_o);
    // top-down FFN: input = lv levels 1..5 + pos
    glom_ff_gemm1<1, true><<<dim3(16, 16, 5), 256, 0, stream>>>(cur, pos_emb, w1td, td_b1, Hbuf);
    glom_ff_gemm2<<<dim3(4, 16, 5), 256, 0, stream>>>(Hbuf, w2td, td_b2, td_o);
    // consensus
    glom_normalize<<<12288, 256, 0, stream>>>(cur, lvbf, kbf);
    glom_sim_gemm<<<dim3(2, 2, 48), 256, 0, stream>>>(lvbf, kbf, sim);
    glom_softmax<<<12288, 256, 0, stream>>>(sim, attn);
    glom_cons_gemm<<<dim3(4, 2, 48), 256, 0, stream>>>(attn, lvbf, cur, tokens, bu_o, td_o, nxt);
    cur = nxt;
  }
}
